// VectorQuantizer_39015482916874
// MI455X (gfx1250) — matrix operand order repaired and checked by reading, not compiled, not run on hardware
//
#include <hip/hip_runtime.h>
#include <hip/hip_bf16.h>

typedef __attribute__((ext_vector_type(16))) __bf16 v16bf;
typedef __attribute__((ext_vector_type(8)))  float  v8f;
typedef int v4i __attribute__((vector_size(16)));   // b128 payload type for async-LDS builtin

#define N_VEC   32768   // B*H*W
#define C_DIM   256
#define K_CODES 1024
#define HW      1024
#define LDSROW  264     // 256 + 8 pad (16B aligned, bank-friendly)

#if defined(__gfx1250__) && \
    __has_builtin(__builtin_amdgcn_global_load_async_to_lds_b128) && \
    __has_builtin(__builtin_amdgcn_s_wait_asynccnt)
#define VQ_ASYNC 1
#define LAS __attribute__((address_space(3)))
#else
#define VQ_ASYNC 0
#endif

__device__ __forceinline__ unsigned short f32_to_bf16(float f) {
    unsigned u = __float_as_uint(f);
    unsigned r = u + 0x7FFFu + ((u >> 16) & 1u);   // round to nearest even
    return (unsigned short)(r >> 16);
}

// ---------------------------------------------------------------------------
// Kernel A: codebook fp32 -> bf16, plus ||e_k||^2.  1024 blocks x 256 threads.
// ---------------------------------------------------------------------------
__global__ void vq_prep_codebook(const float* __restrict__ emb,
                                 unsigned short* __restrict__ emb_bf,
                                 float* __restrict__ enorm) {
    __shared__ float red[256];
    const int k = blockIdx.x;
    const int t = threadIdx.x;
    float v = emb[(size_t)k * C_DIM + t];
    emb_bf[(size_t)k * C_DIM + t] = f32_to_bf16(v);
    red[t] = v * v;
    __syncthreads();
    for (int s = 128; s > 0; s >>= 1) {
        if (t < s) red[t] += red[t + s];
        __syncthreads();
    }
    if (t == 0) enorm[k] = red[0];
}

// ---------------------------------------------------------------------------
// Kernel B: fused GEMM (bf16 WMMA, f32 accum) + argmin over 1024 codes.
// Block = 128 threads (4 waves); wave owns 16 z-rows; 512 blocks total.
// dist_partial(n,k) = ||e_k||^2 - 2 * dot(z_n, e_k)   (||z||^2 constant/row)
// Codebook tiles double-buffered in LDS via GLOBAL_LOAD_ASYNC_TO_LDS_B128
// (ASYNCcnt) when available; synchronous fallback otherwise.
// ---------------------------------------------------------------------------

// stage 16 codes x 256 ch (8 KB bf16) cooperatively: thread -> 64 B
__device__ __forceinline__ void stage_etile_sync(const unsigned short* __restrict__ emb_bf,
                                                 unsigned short* etile, int blk, int tid) {
    const int row  = tid >> 3;
    const int cole = (tid & 7) * 32;
    const uint4* g = (const uint4*)(emb_bf + ((size_t)(blk * 16 + row)) * C_DIM + cole);
    uint4* s = (uint4*)(etile + row * LDSROW + cole);
    s[0] = g[0]; s[1] = g[1]; s[2] = g[2]; s[3] = g[3];
}

#if VQ_ASYNC
__device__ __forceinline__ void stage_etile_async(const unsigned short* __restrict__ emb_bf,
                                                  unsigned short* etile, int blk, int tid) {
    const int row  = tid >> 3;
    const int cole = (tid & 7) * 32;
    v4i* g      = (v4i*)(emb_bf + ((size_t)(blk * 16 + row)) * C_DIM + cole);
    LAS v4i* l  = (LAS v4i*)(etile + row * LDSROW + cole);
    __builtin_amdgcn_global_load_async_to_lds_b128(g, l,  0, 0);
    __builtin_amdgcn_global_load_async_to_lds_b128(g, l, 16, 0);
    __builtin_amdgcn_global_load_async_to_lds_b128(g, l, 32, 0);
    __builtin_amdgcn_global_load_async_to_lds_b128(g, l, 48, 0);
}
#endif

__global__ void __launch_bounds__(128)
vq_argmin_wmma(const float* __restrict__ z,
               const unsigned short* __restrict__ emb_bf,
               const float* __restrict__ enorm,
               int* __restrict__ idx_out,
               float* __restrict__ idx_out_f) {
    __shared__ unsigned short zbf[4][16 * LDSROW];      // per-wave z tile (bf16)
    __shared__ unsigned short etile[2][16 * LDSROW];    // double-buffered codebook tile
    __shared__ float enorm_l[K_CODES];                  // codebook norms

    const int tid  = threadIdx.x;
    const int w    = tid >> 5;          // wave id 0..3
    const int lane = tid & 31;
    const int lr   = lane & 15;         // row/col within 16
    const int hi   = lane >> 4;         // lane half

    // --- stage codebook norms ---
    for (int i = tid; i < K_CODES; i += 128) enorm_l[i] = enorm[i];

    // --- stage this wave's 16x256 z tile as bf16 (z is BCHW; n = b*1024+hw) ---
    const int row0 = blockIdx.x * 64 + w * 16;       // first z-row of this wave
    const int b    = row0 >> 10;
    const int hw0  = row0 & 1023;                    // 16 | 1024 -> no b crossing
    {
        const float* zr = z + (size_t)b * C_DIM * HW + (hw0 + lr);
        const int c0 = hi * 128;
        for (int j = 0; j < 128; ++j) {
            int c = c0 + j;
            zbf[w][lr * LDSROW + c] = f32_to_bf16(zr[(size_t)c * HW]);
        }
    }

    // fragment base pointer for B (ISA 16-bit B 32x16 layout, wave32)
    const unsigned short* bbase = &zbf[w][lr * LDSROW + (hi ? 8 : 0)];
    const int aofs = lr * LDSROW + (hi ? 8 : 0);     // A: codes 16x32 layout

    float runV = 3.0e38f;
    int   runI = 0;

#if VQ_ASYNC
    stage_etile_async(emb_bf, etile[0], 0, tid);
#endif

    for (int blk = 0; blk < 64; ++blk) {
#if VQ_ASYNC
        const unsigned short* abase = etile[blk & 1] + aofs;
        __builtin_amdgcn_s_wait_asynccnt(0);   // our staging of buf[cur] done
        __syncthreads();                       // everyone's done; buf[cur^1] free
        if (blk + 1 < 64)
            stage_etile_async(emb_bf, etile[(blk + 1) & 1], blk + 1, tid);
#else
        const unsigned short* abase = etile[0] + aofs;
        __syncthreads();                       // previous tile fully consumed
        stage_etile_sync(emb_bf, etile[0], blk, tid);
        __syncthreads();
#endif

        v8f acc;
        #pragma unroll
        for (int i = 0; i < 8; ++i) acc[i] = 0.0f;

        #pragma unroll
        for (int kb = 0; kb < 256; kb += 32) {
            union { uint4 q[2]; v16bf v; } A, B;
            A.q[0] = *(const uint4*)(abase + kb);        // K = kb+{0..7}  (+8 if hi)
            A.q[1] = *(const uint4*)(abase + kb + 16);   // K = kb+16+{0..7}
            B.q[0] = *(const uint4*)(bbase + kb);        // K = kb+{0..7}  (+8 if hi)
            B.q[1] = *(const uint4*)(bbase + kb + 16);   // K = kb+16+{0..7}
            acc = __builtin_amdgcn_wmma_f32_16x16x32_bf16(
                      false, A.v, false, B.v, (short)0, acc, false, false);
        }

        // D layout: VGPR r -> code m = r (lanes 0-15) / 8+r (lanes 16-31),
        // column n = lane&15.  Reduce 16 codes of this block for our column.
        float bv = 3.0e38f;
        int   bi = 0;
        const int mofs = blk * 16 + (hi ? 8 : 0);
        #pragma unroll
        for (int r = 0; r < 8; ++r) {
            int   m = mofs + r;
            float d = enorm_l[m] - 2.0f * acc[r];
            if (d < bv) { bv = d; bi = m; }
        }
        // fold the two lane halves (both end up with identical result)
        float ov = __shfl_xor(bv, 16, 32);
        int   oi = __shfl_xor(bi, 16, 32);
        if (ov < bv || (ov == bv && oi < bi)) { bv = ov; bi = oi; }
        if (bv < runV) { runV = bv; runI = bi; }   // blocks ascend -> first-min kept
    }

    if (lane < 16) {
        int n = row0 + lr;
        idx_out[n]   = runI;
        idx_out_f[n] = (float)runI;
    }
}

// ---------------------------------------------------------------------------
// Kernel C: epilogue.  zq (BCHW) and loss = 1.25*(zq - z)^2 (BHWC), fp32.
// Tile 64 c x 32 hw with LDS transpose so every global access is coalesced.
// 4096 blocks x 256 threads.
// ---------------------------------------------------------------------------
__global__ void __launch_bounds__(256)
vq_epilogue(const float* __restrict__ z,
            const float* __restrict__ emb,
            const int* __restrict__ idx,
            float* __restrict__ zq_out,
            float* __restrict__ loss_out) {
    __shared__ float tile[64 * 33];    // [c_local][hw_local], padded

    const int t      = threadIdx.x;
    const int wgid   = blockIdx.x;
    const int b      = wgid >> 7;
    const int t2     = wgid & 127;
    const int cbase  = (t2 >> 5) * 64;
    const int hwbase = (t2 & 31) * 32;

    // read phase: coalesced over hw (BCHW)
    {
        const int hw_l = t & 31;
        const int c0   = t >> 5;              // 0..7
        const int hw   = hwbase + hw_l;
        const int n    = b * HW + hw;
        const int k    = idx[n];
        #pragma unroll
        for (int j = 0; j < 8; ++j) {
            int c_local = c0 + j * 8;
            int c = cbase + c_local;
            size_t zi = ((size_t)(b * C_DIM + c)) * HW + hw;
            float zv = z[zi];
            float ev = emb[(size_t)k * C_DIM + c];
            zq_out[zi] = ev;                           // straight-through fwd = e[idx]
            float d = ev - zv;
            tile[c_local * 33 + hw_l] = 1.25f * d * d; // 0.25*commit + 1.0*codebook
        }
    }
    __syncthreads();

    // write phase: coalesced over c (BHWC)
    {
        const int c_l   = t & 63;
        const int hw_l0 = t >> 6;             // 0..3
        #pragma unroll
        for (int j = 0; j < 8; ++j) {
            int hw_l = hw_l0 + j * 4;
            int n = b * HW + hwbase + hw_l;
            loss_out[(size_t)n * C_DIM + cbase + c_l] = tile[c_l * 33 + hw_l];
        }
    }
}

// ---------------------------------------------------------------------------
extern "C" void kernel_launch(void* const* d_in, const int* in_sizes, int n_in,
                              void* d_out, int out_size, void* d_ws, size_t ws_size,
                              hipStream_t stream) {
    const float* z   = (const float*)d_in[0];   // [32,256,32,32]
    const float* emb = (const float*)d_in[1];   // [1024,256]

    float* out   = (float*)d_out;
    float* zq    = out;                              // 8,388,608  (BCHW)
    float* loss  = out + (size_t)N_VEC * C_DIM;      // 8,388,608  (BHWC)
    float* idx_f = loss + (size_t)N_VEC * C_DIM;     // 32,768

    // workspace layout
    unsigned char* ws = (unsigned char*)d_ws;
    unsigned short* emb_bf = (unsigned short*)ws;                        // 512 KB
    float*          enorm  = (float*)(ws + (size_t)K_CODES * C_DIM * 2); // 4 KB
    int*            idx_i  = (int*)(ws + (size_t)K_CODES * C_DIM * 2 + K_CODES * 4); // 128 KB

    vq_prep_codebook<<<K_CODES, 256, 0, stream>>>(emb, emb_bf, enorm);
    vq_argmin_wmma<<<N_VEC / 64, 128, 0, stream>>>(z, emb_bf, enorm, idx_i, idx_f);
    vq_epilogue<<<(N_VEC / 32) * (C_DIM / 64), 256, 0, stream>>>(z, emb, idx_i, zq, loss);
}
